// CCG_83176336654934
// MI455X (gfx1250) — matrix operand order repaired and checked by reading, not compiled, not run on hardware
//
#include <hip/hip_runtime.h>
#include <cstdint>
#include <cstddef>

// ---------------- problem constants ----------------
#define Bc   128
#define Lc   256
#define Ec   300
#define EP   320      // E padded to multiple of 32 (WMMA K)
#define Hc   512
#define G4   2048     // 4*H
#define Tc   9
#define TP   16       // T padded
#define H2   1024     // 2*H
// padded LDS row strides (halfs) to dodge bank conflicts
#define XS   328
#define HS   520

typedef __attribute__((ext_vector_type(16))) _Float16 v16h;
typedef __attribute__((ext_vector_type(8)))  float    v8f;

__device__ __forceinline__ float sigf(float x) { return 1.0f / (1.0f + __expf(-x)); }

// A-matrix fragment (16-bit, 16x32). Per ISA 7.12.2:
// lane(0..15)=row M, lane>=16 same M other K-half.
// VGPR v holds K pair: k0 = (v>>2)*16 + (lane>>4)*8 + (v&3)*2
__device__ __forceinline__ v16h load_a(const _Float16* __restrict__ base,
                                       int rowStride, int kbase, int lane) {
  const int m  = lane & 15;
  const int kh = lane >> 4;
  union { v16h v; uint32_t u[8]; } r;
  const _Float16* p = base + m * rowStride + kbase;
#pragma unroll
  for (int v = 0; v < 8; ++v) {
    const int k0 = ((v >> 2) << 4) + (kh << 3) + ((v & 3) << 1);
    r.u[v] = *(const uint32_t*)(p + k0);
  }
  return r.v;
}

// B-matrix fragment (16-bit, 32x16), element (k,n) at W[n*rowStride + k]
// lane: n = lane&15 ; K pair per VGPR v: k = (v>>2)*16 + (lane>>4)*8 + (v&3)*2
__device__ __forceinline__ v16h load_b(const _Float16* __restrict__ W,
                                       int rowStride, int nbase, int kbase, int lane) {
  const int n  = nbase + (lane & 15);
  const int kh = lane >> 4;
  union { v16h v; uint32_t u[8]; } r;
  const _Float16* p = W + (size_t)n * rowStride + kbase + (kh << 3);
#pragma unroll
  for (int v = 0; v < 8; ++v) r.u[v] = *(const uint32_t*)(p + ((v >> 2) << 4) + ((v & 3) << 1));
  return r.v;
}

// ---------------- async LDS<->global helpers (gfx1250) ----------------
__device__ __forceinline__ void wait_async0() {
  asm volatile("s_wait_asynccnt 0x0" ::: "memory");
}

// stage x tile [16 rows][EP halfs] for timestep lt into xbuf (stride XS),
// 1280 b64 chunks spread over 256 threads -> 5 per thread
__device__ __forceinline__ void issue_x_async(const _Float16* __restrict__ emb,
                                              int bb0, int lt,
                                              _Float16* xbuf, int tid) {
#pragma unroll
  for (int it = 0; it < 5; ++it) {
    const int c  = tid + it * 256;            // 0..1279
    const int m  = c / 80;
    const int kk = (c - m * 80) * 4;          // half index, 8-byte chunks
    const uint32_t lds = (uint32_t)(uintptr_t)(xbuf + m * XS + kk);
    const unsigned long long g =
        (unsigned long long)(uintptr_t)(emb + ((size_t)(bb0 + m) * Lc + lt) * EP + kk);
    asm volatile("global_load_async_to_lds_b64 %0, %1, off"
                 :: "v"(lds), "v"(g) : "memory");
  }
}

// write h tile [16 rows][Hc halfs] (stride HS) to hout[b][l][dir*Hc + c],
// 1024 b128 chunks over 256 threads -> 4 per thread
__device__ __forceinline__ void issue_h_store(const _Float16* __restrict__ hn,
                                              _Float16* __restrict__ hout,
                                              int bb0, int lt, int dir, int tid) {
#pragma unroll
  for (int it = 0; it < 4; ++it) {
    const int c  = tid + it * 256;            // 0..1023
    const int m  = c >> 6;
    const int kk = (c & 63) * 8;              // half index, 16-byte chunks
    const uint32_t lds = (uint32_t)(uintptr_t)(hn + m * HS + kk);
    const unsigned long long g =
        (unsigned long long)(uintptr_t)(hout + ((size_t)(bb0 + m) * Lc + lt) * H2 +
                                        dir * Hc + kk);
    asm volatile("global_store_async_from_lds_b128 %0, %1, off"
                 :: "v"(g), "v"(lds) : "memory");
  }
}

// ---------------- prep kernels ----------------
__global__ void cvt_pad(const float* __restrict__ src, _Float16* __restrict__ dst,
                        int srcRows, int srcCols, int dstCols, int total) {
  int i = blockIdx.x * blockDim.x + threadIdx.x;
  if (i >= total) return;
  int r = i / dstCols, c = i - r * dstCols;
  dst[i] = (r < srcRows && c < srcCols) ? (_Float16)src[(size_t)r * srcCols + c]
                                        : (_Float16)0.0f;
}

__global__ void bias_sum(const float* __restrict__ a, const float* __restrict__ b,
                         float* __restrict__ o, int n) {
  int i = blockIdx.x * blockDim.x + threadIdx.x;
  if (i < n) o[i] = a[i] + b[i];
}

__global__ void embed_k(const int* __restrict__ ids, const float* __restrict__ table,
                        _Float16* __restrict__ out, int total) {
  int i = blockIdx.x * blockDim.x + threadIdx.x;
  if (i >= total) return;
  int bl = i / EP, k = i - bl * EP;
  int id = ids[bl];
  out[i] = (k < Ec) ? (_Float16)table[(size_t)id * Ec + k] : (_Float16)0.0f;
}

// ---------------- fused LSTM recurrence ----------------
// grid (8, 2): x = batch tile of 16 rows, y = direction. block = 256 (8 waves).
// per step: g = x_t @ Wih^T + h @ Whh^T + bias ; gates ; c,h update.
// x tiles double-buffered via ASYNC loads issued one step ahead; h written back
// via ASYNC stores from LDS. c-state lives in registers (4x v8f per wave).
__global__ void __launch_bounds__(256) lstm_rec(
    const _Float16* __restrict__ emb,      // [B*L][EP]
    const _Float16* __restrict__ WihF,     // [G4][EP]
    const _Float16* __restrict__ WihB,
    const _Float16* __restrict__ WhhF,     // [G4][Hc]
    const _Float16* __restrict__ WhhB,
    const float*    __restrict__ biasF,    // [G4]
    const float*    __restrict__ biasB,
    _Float16*       __restrict__ hout)     // [B*L][H2]
{
  __shared__ _Float16 x_lds[2][16 * XS];
  __shared__ _Float16 h_lds[2][16 * HS];

  const int tid  = threadIdx.x;
  const int lane = tid & 31;
  const int wave = tid >> 5;
  const int dir  = blockIdx.y;
  const int bb0  = blockIdx.x * 16;

  const _Float16* Wih  = dir ? WihB  : WihF;
  const _Float16* Whh  = dir ? WhhB  : WhhF;
  const float*    bias = dir ? biasB : biasF;

  // zero both h buffers; kick off async load of first x tile
  for (int i = tid; i < 2 * 16 * HS; i += 256)
    ((_Float16*)h_lds)[i] = (_Float16)0.0f;
  issue_x_async(emb, bb0, dir ? (Lc - 1) : 0, x_lds[0], tid);
  v8f cst[4] = {};

  int cur = 0, xb = 0;
  for (int t = 0; t < Lc; ++t) {
    const int lt = dir ? (Lc - 1 - t) : t;

    wait_async0();       // this thread's x tile + prior h stores complete
    __syncthreads();     // all threads' async writes to LDS visible

    // prefetch next step's x tile into the other buffer (overlaps GEMM)
    const int tn  = (t + 1 < Lc) ? (t + 1) : t;
    const int ltn = dir ? (Lc - 1 - tn) : tn;
    issue_x_async(emb, bb0, ltn, x_lds[xb ^ 1], tid);

    const _Float16* xc = x_lds[xb];
    const _Float16* hc = h_lds[cur];
    _Float16*       hn = h_lds[cur ^ 1];

#pragma unroll
    for (int qi = 0; qi < 4; ++qi) {
      const int q = (wave << 2) + qi;
      v8f acc[4] = {};
      // input-projection phase, K = EP
      for (int kb = 0; kb < EP; kb += 32) {
        v16h a = load_a(xc, XS, kb, lane);
#pragma unroll
        for (int gi = 0; gi < 4; ++gi) {
          v16h b = load_b(Wih, EP, gi * Hc + (q << 4), kb, lane);
          acc[gi] = __builtin_amdgcn_wmma_f32_16x16x32_f16(
              false, a, false, b, (short)0, acc[gi], false, false);
        }
      }
      // recurrent phase, K = Hc
      for (int kb = 0; kb < Hc; kb += 32) {
        v16h a = load_a(hc, HS, kb, lane);
#pragma unroll
        for (int gi = 0; gi < 4; ++gi) {
          v16h b = load_b(Whh, Hc, gi * Hc + (q << 4), kb, lane);
          acc[gi] = __builtin_amdgcn_wmma_f32_16x16x32_f16(
              false, a, false, b, (short)0, acc[gi], false, false);
        }
      }
      // gate math: lane's 8 elems share column n, rows m = j + 8*(lane>>4)
      const int n  = (q << 4) + (lane & 15);
      const float bi = bias[n], bf = bias[Hc + n], bg = bias[2 * Hc + n], bo = bias[3 * Hc + n];
#pragma unroll
      for (int j = 0; j < 8; ++j) {
        float iv = sigf(acc[0][j] + bi);
        float fv = sigf(acc[1][j] + bf);
        float gv = tanhf(acc[2][j] + bg);
        float ov = sigf(acc[3][j] + bo);
        float c  = fv * cst[qi][j] + iv * gv;
        cst[qi][j] = c;
        float hv = ov * tanhf(c);
        const int m = j + ((lane >> 4) << 3);
        hn[m * HS + n] = (_Float16)hv;   // next-step buffer: no read/write overlap
      }
    }
    __syncthreads();   // h_lds[nxt] complete across block

    // async writeback of h tile from LDS to global (f16)
    issue_h_store(hn, hout, bb0, lt, dir, tid);

    cur ^= 1;
    xb  ^= 1;
  }
  wait_async0();       // drain stores (S_ENDPGM also wait-idles)
}

// ---------------- emissions GEMM (WMMA, T padded to 16) ----------------
// grid 256, block 256 (8 waves); each wave: 16 bl-rows x 16 tags, K=1024
__global__ void __launch_bounds__(256) emis_k(
    const _Float16* __restrict__ hout,   // [B*L][H2]
    const _Float16* __restrict__ Wem16,  // [TP][H2] zero padded rows
    const float*    __restrict__ bem,    // [Tc]
    float*          __restrict__ ems)    // [B*L][TP]
{
  const int lane = threadIdx.x & 31;
  const int wave = threadIdx.x >> 5;
  const size_t row0 = ((size_t)blockIdx.x * 8 + wave) * 16;
  const _Float16* A = hout + row0 * H2;
  v8f acc = {};
  for (int kb = 0; kb < H2; kb += 32) {
    v16h a = load_a(A, H2, kb, lane);
    v16h b = load_b(Wem16, H2, 0, kb, lane);
    acc = __builtin_amdgcn_wmma_f32_16x16x32_f16(
        false, a, false, b, (short)0, acc, false, false);
  }
  const int n = lane & 15;
  const float bb = (n < Tc) ? bem[n] : 0.0f;
#pragma unroll
  for (int j = 0; j < 8; ++j) {
    const int m = j + ((lane >> 4) << 3);
    ems[(row0 + m) * TP + n] = acc[j] + bb;
  }
}

// ---------------- CRF score + partition + reduce ----------------
__global__ void __launch_bounds__(128) crf_k(
    const float* __restrict__ ems,       // [B*L][TP]
    const int* __restrict__ labels,      // [B][L]
    const unsigned char* __restrict__ masks, // [B][L] bool
    const float* __restrict__ start_t, const float* __restrict__ end_t,
    const float* __restrict__ trans,    // [T][T]
    float* __restrict__ out)
{
  __shared__ float tr[Tc * Tc], st[Tc], et[Tc], red[128];
  const int b = threadIdx.x;
  if (b < Tc * Tc) tr[b] = trans[b];
  if (b < Tc) { st[b] = start_t[b]; et[b] = end_t[b]; }
  __syncthreads();

  const float* em = ems + (size_t)b * Lc * TP;
  const int* tg = labels + b * Lc;
  const unsigned char* mk = masks + b * Lc;

  // gold path score
  int prev = tg[0];
  float score = st[prev] + em[prev];
  int cnt = mk[0] ? 1 : 0;
  for (int l = 1; l < Lc; ++l) {
    const int curt = tg[l];
    const float m = mk[l] ? 1.0f : 0.0f;
    score += (tr[prev * Tc + curt] + em[l * TP + curt]) * m;
    cnt += mk[l] ? 1 : 0;
    prev = curt;
  }
  score += et[tg[cnt - 1]];

  // forward algorithm (log partition)
  float alpha[Tc];
#pragma unroll
  for (int i = 0; i < Tc; ++i) alpha[i] = st[i] + em[i];
  for (int l = 1; l < Lc; ++l) {
    float na[Tc];
#pragma unroll
    for (int n = 0; n < Tc; ++n) {
      float mx = -1e30f;
#pragma unroll
      for (int m2 = 0; m2 < Tc; ++m2) mx = fmaxf(mx, alpha[m2] + tr[m2 * Tc + n]);
      float s = 0.0f;
#pragma unroll
      for (int m2 = 0; m2 < Tc; ++m2) s += __expf(alpha[m2] + tr[m2 * Tc + n] - mx);
      na[n] = mx + __logf(s) + em[l * TP + n];
    }
    const bool mm = mk[l] != 0;
#pragma unroll
    for (int n = 0; n < Tc; ++n) alpha[n] = mm ? na[n] : alpha[n];
  }
  float mx = -1e30f;
#pragma unroll
  for (int n = 0; n < Tc; ++n) mx = fmaxf(mx, alpha[n] + et[n]);
  float s = 0.0f;
#pragma unroll
  for (int n = 0; n < Tc; ++n) s += __expf(alpha[n] + et[n] - mx);
  const float logZ = mx + __logf(s);

  red[b] = logZ - score;   // -llh contribution
  __syncthreads();
  if (b == 0) {
    float acc = 0.0f;
    for (int i = 0; i < 128; ++i) acc += red[i];
    out[0] = acc;
  }
}

// ---------------- host launcher ----------------
extern "C" void kernel_launch(void* const* d_in, const int* in_sizes, int n_in,
                              void* d_out, int out_size, void* d_ws, size_t ws_size,
                              hipStream_t stream) {
  const int*   ids    = (const int*)d_in[0];
  const unsigned char* masks = (const unsigned char*)d_in[1];
  const int*   labels = (const int*)d_in[2];
  const float* table  = (const float*)d_in[3];
  const float* Wih_f  = (const float*)d_in[4];
  const float* Whh_f  = (const float*)d_in[5];
  const float* bih_f  = (const float*)d_in[6];
  const float* bhh_f  = (const float*)d_in[7];
  const float* Wih_b  = (const float*)d_in[8];
  const float* Whh_b  = (const float*)d_in[9];
  const float* bih_b  = (const float*)d_in[10];
  const float* bhh_b  = (const float*)d_in[11];
  const float* W_em   = (const float*)d_in[12];
  const float* b_em   = (const float*)d_in[13];
  const float* start_t= (const float*)d_in[14];
  const float* end_t  = (const float*)d_in[15];
  const float* trans  = (const float*)d_in[16];

  // carve workspace
  char* ws = (char*)d_ws;
  size_t off = 0;
  auto carve = [&](size_t bytes) { char* p = ws + off; off = (off + bytes + 255) & ~(size_t)255; return p; };
  _Float16* wihf16f = (_Float16*)carve((size_t)G4 * EP * 2);
  _Float16* wihf16b = (_Float16*)carve((size_t)G4 * EP * 2);
  _Float16* whhf16f = (_Float16*)carve((size_t)G4 * Hc * 2);
  _Float16* whhf16b = (_Float16*)carve((size_t)G4 * Hc * 2);
  _Float16* wem16   = (_Float16*)carve((size_t)TP * H2 * 2);
  float*    biasf   = (float*)carve((size_t)G4 * 4);
  float*    biasb   = (float*)carve((size_t)G4 * 4);
  _Float16* emb     = (_Float16*)carve((size_t)Bc * Lc * EP * 2);
  _Float16* hout    = (_Float16*)carve((size_t)Bc * Lc * H2 * 2);
  float*    ems     = (float*)carve((size_t)Bc * Lc * TP * 4);

  const int TB = 256;
  auto blks = [](int n, int tb) { return (n + tb - 1) / tb; };

  // weight conversion / padding
  int nIh = G4 * EP;
  cvt_pad<<<blks(nIh, TB), TB, 0, stream>>>(Wih_f, wihf16f, G4, Ec, EP, nIh);
  cvt_pad<<<blks(nIh, TB), TB, 0, stream>>>(Wih_b, wihf16b, G4, Ec, EP, nIh);
  int nHh = G4 * Hc;
  cvt_pad<<<blks(nHh, TB), TB, 0, stream>>>(Whh_f, whhf16f, G4, Hc, Hc, nHh);
  cvt_pad<<<blks(nHh, TB), TB, 0, stream>>>(Whh_b, whhf16b, G4, Hc, Hc, nHh);
  int nEm = TP * H2;
  cvt_pad<<<blks(nEm, TB), TB, 0, stream>>>(W_em, wem16, Tc, H2, H2, nEm);
  bias_sum<<<blks(G4, TB), TB, 0, stream>>>(bih_f, bhh_f, biasf, G4);
  bias_sum<<<blks(G4, TB), TB, 0, stream>>>(bih_b, bhh_b, biasb, G4);

  // embedding gather (padded)
  int nEmb = Bc * Lc * EP;
  embed_k<<<blks(nEmb, TB), TB, 0, stream>>>(ids, table, emb, nEmb);

  // fused bidirectional LSTM recurrence (async-LDS staged)
  lstm_rec<<<dim3(Bc / 16, 2), 256, 0, stream>>>(emb, wihf16f, wihf16b,
                                                 whhf16f, whhf16b, biasf, biasb, hout);

  // emissions GEMM
  emis_k<<<(Bc * Lc) / (16 * 8), 256, 0, stream>>>(hout, wem16, b_em, ems);

  // CRF + reduction -> scalar NLL
  crf_k<<<1, 128, 0, stream>>>(ems, labels, masks, start_t, end_t, trans, (float*)d_out);
}
